// AdaAttN_37580963840521
// MI455X (gfx1250) — matrix operand order repaired and checked by reading, not compiled, not run on hardware
//
#include <hip/hip_runtime.h>
#include <hip/hip_bf16.h>

typedef __attribute__((ext_vector_type(16))) _Float16 v16h;
typedef __attribute__((ext_vector_type(8)))  _Float16 v8h;
typedef __attribute__((ext_vector_type(8)))  float    v8f;
typedef __attribute__((ext_vector_type(4)))  unsigned int u32x4;
typedef __attribute__((ext_vector_type(8)))  int      i32x8;
typedef __attribute__((ext_vector_type(4)))  int      i32x4;

#define B_  4
#define C_  512
#define NT  4096   // 64*64 tokens per image

// ---------------------------------------------------------------- WMMA utils

__device__ __forceinline__ v8f wmma_f32_f16(v16h a, v16h b, v8f c) {
  // D = A(16x32 f16) * B(32x16 f16) + C(16x16 f32)
  return __builtin_amdgcn_wmma_f32_16x16x32_f16(false, a, false, b, (short)0, c,
                                                false, false);
}

// A fragment: 16x32 f16 from row-major storage (row = M index), ld in halfs.
// Lane h*16+m holds row m, K in [16h, 16h+16).
__device__ __forceinline__ v16h load_frag_a(const _Float16* base, int ld, int lane) {
  const int h = lane >> 4, m = lane & 15;
  const _Float16* p = base + m * ld + 16 * h;
  v8h lo = *(const v8h*)(p);
  v8h hi = *(const v8h*)(p + 8);
  v16h r;
#pragma unroll
  for (int i = 0; i < 8; ++i) { r[i] = lo[i]; r[8 + i] = hi[i]; }
  return r;
}

// B fragment: 32x16 f16; storage row n = output column (N index), ld in halfs.
// Lane h*16+n holds column n, K in [16h, 16h+16) -> one contiguous 32B read.
__device__ __forceinline__ v16h load_frag_b(const _Float16* base, int ld, int lane) {
  const int h = lane >> 4, n = lane & 15;
  return *(const v16h*)(base + n * ld + 16 * h);
}

// ------------------------------------------------- TDM: global -> LDS tile
// D# per CDNA5 ISA ch.8: group0 = {count, lds_addr, global_addr, type=2};
// group1 = {mask/flags/data_size, tensor_dim0/1, tile_dim0/1/2, strides}.
// data_size = 3 (8-byte units). tile_d1==0 -> 1-D copy.
__device__ __forceinline__ void tdm_load(unsigned lds_off, const void* gsrc,
                                         unsigned tile_d0, unsigned tile_d1,
                                         unsigned tensor_d1, unsigned stride0) {
  unsigned long long ga = (unsigned long long)(size_t)gsrc;
  u32x4 g0;
  g0[0] = 1u;                                         // count = 1, user mode
  g0[1] = lds_off;                                    // LDS byte address
  g0[2] = (unsigned)ga;                               // global_addr[31:0]
  g0[3] = (unsigned)((ga >> 32) & 0x1ffffffu) | 0x80000000u;  // addr[56:32] | type=2
  i32x8 g1;
  g1[0] = (int)(3u << 16);                            // wg_mask=0, data_size=8B
  g1[1] = (int)((tile_d0 & 0xffffu) << 16);           // tensor_dim0 lo16 (== tile_d0)
  g1[2] = (int)((tile_d0 >> 16) | ((tensor_d1 & 0xffffu) << 16));  // td0 hi | td1 lo
  g1[3] = (int)((tensor_d1 >> 16) | (tile_d0 << 16)); // td1 hi | tile_dim0
  g1[4] = (int)(tile_d1 & 0xffffu);                   // tile_dim1 (0 => 1-D), tile_dim2=0
  g1[5] = (int)stride0;                               // tensor_dim0_stride lo32
  g1[6] = 0;                                          // stride0 hi16 | dim1_stride lo16
  g1[7] = 0;
  i32x4 z4 = {0, 0, 0, 0};
  i32x8 z8 = {0, 0, 0, 0, 0, 0, 0, 0};
  __builtin_amdgcn_tensor_load_to_lds(g0, g1, z4, z4, z8, 0);
}

// -------------------------------------------------- 1) transpose + f32->f16
// in:  [C_][NT] f32 (one batch selected by blockIdx.z);  out: [NT][C_] f16
__global__ __launch_bounds__(256) void k_transpose_cvt(const float* __restrict__ in,
                                                       _Float16* __restrict__ out) {
  __shared__ float tile[32][33];
  const int b = blockIdx.z;
  const int tbase = blockIdx.x * 32;
  const int cbase = blockIdx.y * 32;
  const float* src = in + ((size_t)b * C_ + cbase) * NT + tbase;
#pragma unroll
  for (int i = 0; i < 4; ++i) {
    int c = threadIdx.y + i * 8;
    tile[c][threadIdx.x] = src[(size_t)c * NT + threadIdx.x];
  }
  __syncthreads();
  _Float16* dst = out + ((size_t)b * NT + tbase) * C_ + cbase;
#pragma unroll
  for (int i = 0; i < 4; ++i) {
    int t = threadIdx.y + i * 8;
    dst[(size_t)t * C_ + threadIdx.x] = (_Float16)tile[threadIdx.x][t];
  }
}

// -------------------------------------------------- 2) weights f32 -> f16
__global__ __launch_bounds__(256) void k_cvt_w(const float* __restrict__ in,
                                               _Float16* __restrict__ out, int n) {
  int i = blockIdx.x * 256 + threadIdx.x;
  if (i < n) out[i] = (_Float16)in[i];
}

// -------------------------------------------------- 3) instance-norm stats
__global__ __launch_bounds__(256) void k_inorm_stats(const float* __restrict__ x,
                                                     float* __restrict__ mu,
                                                     float* __restrict__ rstd) {
  const int bc = blockIdx.x;
  const float* p = x + (size_t)bc * NT;
  float s = 0.f, ss = 0.f;
  for (int i = threadIdx.x; i < NT; i += 256) {
    float v = p[i];
    s += v; ss += v * v;
  }
#pragma unroll
  for (int o = 16; o > 0; o >>= 1) {
    s  += __shfl_down(s, o, 32);
    ss += __shfl_down(ss, o, 32);
  }
  __shared__ float as_[8], ass_[8];
  const int wid = threadIdx.x >> 5, ln = threadIdx.x & 31;
  if (ln == 0) { as_[wid] = s; ass_[wid] = ss; }
  __syncthreads();
  if (threadIdx.x == 0) {
    float S = 0.f, SS = 0.f;
#pragma unroll
    for (int i = 0; i < 8; ++i) { S += as_[i]; SS += ass_[i]; }
    float mean = S * (1.f / NT);
    float var  = (SS - S * mean) * (1.f / (NT - 1));   // Bessel ddof=1
    float sd   = sqrtf(fmaxf(var, 0.f)) + 1e-6f;
    mu[bc]   = mean;
    rstd[bc] = 1.f / sd;
  }
}

// -------------------------------------------------- 4) projection GEMMs
__global__ __launch_bounds__(128) void k_proj(const _Float16* __restrict__ xh,
                                              const _Float16* __restrict__ wh,
                                              const float* __restrict__ bias,
                                              _Float16* __restrict__ yrow,
                                              _Float16* __restrict__ vt,
                                              _Float16* __restrict__ v2t,
                                              int mode) {
  const int lane = threadIdx.x & 31;
  const int wid  = threadIdx.x >> 5;
  const int tt   = blockIdx.x * 4 + wid;          // 0 .. B_*256-1
  const int b    = tt >> 8;
  const int tbase = (tt & 255) * 16;
  const int obase = blockIdx.y * 128;

  const _Float16* X = xh + ((size_t)b * NT + tbase) * C_;
  v8f acc[8];
  v8f zero = {};
#pragma unroll
  for (int ot = 0; ot < 8; ++ot) acc[ot] = zero;

  for (int kc = 0; kc < C_ / 32; ++kc) {
    v16h a = load_frag_a(X + kc * 32, C_, lane);
#pragma unroll
    for (int ot = 0; ot < 8; ++ot) {
      v16h bf = load_frag_b(wh + (size_t)(obase + ot * 16) * C_ + kc * 32, C_, lane);
      acc[ot] = wmma_f32_f16(a, bf, acc[ot]);
    }
  }

  const int h = lane >> 4, n = lane & 15;
#pragma unroll
  for (int ot = 0; ot < 8; ++ot) {
    const int o = obase + ot * 16 + n;
    const float bv = bias[o];
    if (mode == 0) {
#pragma unroll
      for (int r = 0; r < 8; ++r) {
        yrow[((size_t)b * NT + tbase + r + 8 * h) * C_ + o] =
            (_Float16)(acc[ot][r] + bv);
      }
    } else {
      v8h pk, pk2;
#pragma unroll
      for (int r = 0; r < 8; ++r) {
        float v = acc[ot][r] + bv;
        pk[r]  = (_Float16)v;
        pk2[r] = (_Float16)(v * v);
      }
      size_t addr = ((size_t)b * C_ + o) * NT + tbase + 8 * h;
      *(v8h*)(vt  + addr) = pk;
      *(v8h*)(v2t + addr) = pk2;
    }
  }
}

// -------------------------------------------------- 5) fused flash attention
// 4 waves/block = 4 consecutive query tiles, same batch + channel half.
// K/V/V2 32-key tiles staged to LDS by the Tensor Data Mover, double-buffered;
// TDM for tile i+1 overlaps WMMA on tile i. One barrier per iteration.
__global__ __launch_bounds__(128) void k_attn(const _Float16* __restrict__ Qh,
                                              const _Float16* __restrict__ Kh,
                                              const _Float16* __restrict__ Vt,
                                              const _Float16* __restrict__ V2t,
                                              const float* __restrict__ content,
                                              const float* __restrict__ mu,
                                              const float* __restrict__ rstd,
                                              float* __restrict__ out) {
  __shared__ __align__(128) _Float16 Klds [2][32 * 512];   // 64 KB
  __shared__ __align__(128) _Float16 Vlds [2][256 * 32];   // 32 KB
  __shared__ __align__(128) _Float16 V2lds[2][256 * 32];   // 32 KB
  __shared__ __align__(128) _Float16 Plds [4][16 * 32];    //  4 KB

  const int lane  = threadIdx.x & 31;
  const int wid   = threadIdx.x >> 5;
  const int qt    = blockIdx.x * 4 + wid;          // 0 .. B_*256-1 (same b per block)
  const int b     = qt >> 8;
  const int qbase = (qt & 255) * 16;
  const int cbase = blockIdx.y * 256;
  const int h = lane >> 4, n = lane & 15;

  const _Float16* Q  = Qh  + ((size_t)b * NT + qbase) * C_;
  const _Float16* K  = Kh  + (size_t)b * NT * C_;
  const _Float16* V  = Vt  + ((size_t)b * C_ + cbase) * NT;
  const _Float16* V2 = V2t + ((size_t)b * C_ + cbase) * NT;

  // Resident Q fragments (16 K-chunks of 32)
  v16h qf[16];
#pragma unroll
  for (int kc = 0; kc < 16; ++kc) qf[kc] = load_frag_a(Q + kc * 32, C_, lane);

  v8f zero = {};
  v8f accM[16], accS[16];
#pragma unroll
  for (int ct = 0; ct < 16; ++ct) { accM[ct] = zero; accS[ct] = zero; }

  float mrun[8], srun[8];
#pragma unroll
  for (int r = 0; r < 8; ++r) { mrun[r] = -3.0e38f; srun[r] = 0.f; }

  // Prime buffer 0 with key tile kb=0 (uniform addresses; wave 0 issues TDM).
  if (wid == 0) {
    tdm_load((unsigned)(size_t)&Klds[0][0],  K,  4096, 0,   1,   4096); // 32KB 1-D
    tdm_load((unsigned)(size_t)&Vlds[0][0],  V,  8,    256, 256, 1024); // 64B x 256 rows
    tdm_load((unsigned)(size_t)&V2lds[0][0], V2, 8,    256, 256, 1024);
  }

  int it = 0;
#pragma unroll 1
  for (int kb = 0; kb < NT; kb += 32, it ^= 1) {
    if (wid == 0) __builtin_amdgcn_s_wait_tensorcnt(0);  // buffer `it` ready
    __syncthreads();   // also proves buffer it^1 fully consumed by all waves
    if (wid == 0 && kb + 32 < NT) {
      const int nxt = it ^ 1;
      tdm_load((unsigned)(size_t)&Klds[nxt][0],  K  + (size_t)(kb + 32) * C_,
               4096, 0, 1, 4096);
      tdm_load((unsigned)(size_t)&Vlds[nxt][0],  V  + (kb + 32), 8, 256, 256, 1024);
      tdm_load((unsigned)(size_t)&V2lds[nxt][0], V2 + (kb + 32), 8, 256, 256, 1024);
    }

    const _Float16* Kt  = &Klds[it][0];
    const _Float16* Vb  = &Vlds[it][0];
    const _Float16* V2b = &V2lds[it][0];

    // ---- logits 16x32 (two 16x16 D tiles) from LDS
    v8f L0 = zero, L1 = zero;
#pragma unroll
    for (int kc = 0; kc < 16; ++kc) {
      v16h b0 = load_frag_b(Kt + kc * 32, C_, lane);                   // keys 0..15
      v16h b1 = load_frag_b(Kt + (size_t)16 * C_ + kc * 32, C_, lane); // keys 16..31
      L0 = wmma_f32_f16(qf[kc], b0, L0);
      L1 = wmma_f32_f16(qf[kc], b1, L1);
    }

    // ---- online softmax row stats + P -> LDS (f16)
    float f[8];
#pragma unroll
    for (int r = 0; r < 8; ++r) {
      float mx = fmaxf(L0[r], L1[r]);
#pragma unroll
      for (int m = 1; m <= 8; m <<= 1) mx = fmaxf(mx, __shfl_xor(mx, m, 32));
      float mnew = fmaxf(mrun[r], mx);
      f[r] = __expf(mrun[r] - mnew);
      float e0 = __expf(L0[r] - mnew);
      float e1 = __expf(L1[r] - mnew);
      float rs = e0 + e1;
#pragma unroll
      for (int m = 1; m <= 8; m <<= 1) rs += __shfl_xor(rs, m, 32);
      srun[r] = srun[r] * f[r] + rs;
      mrun[r] = mnew;
      Plds[wid][(r + 8 * h) * 32 + n]      = (_Float16)e0;
      Plds[wid][(r + 8 * h) * 32 + 16 + n] = (_Float16)e1;
    }
    v8f fvec;
#pragma unroll
    for (int r = 0; r < 8; ++r) fvec[r] = f[r];

    v16h pa = load_frag_a(&Plds[wid][0], 32, lane);  // P as A fragment (16x32)

#pragma unroll
    for (int ct = 0; ct < 16; ++ct) {
      accM[ct] *= fvec;
      accS[ct] *= fvec;
    }
#pragma unroll
    for (int ct = 0; ct < 16; ++ct) {
      v16h bv  = load_frag_b(Vb  + (size_t)(ct * 16) * 32, 32, lane);
      accM[ct] = wmma_f32_f16(pa, bv, accM[ct]);
      v16h bv2 = load_frag_b(V2b + (size_t)(ct * 16) * 32, 32, lane);
      accS[ct] = wmma_f32_f16(pa, bv2, accS[ct]);
    }
  }

  // ---- epilogue: normalize, std = sqrt(relu(m2 - mean^2)), AdaAttN blend
  v8f rsv;
#pragma unroll
  for (int r = 0; r < 8; ++r) rsv[r] = 1.f / srun[r];

#pragma unroll
  for (int ct = 0; ct < 16; ++ct) {
    const int c = cbase + ct * 16 + n;
    const float mu_ = mu[b * C_ + c];
    const float rs_ = rstd[b * C_ + c];
    const size_t base = ((size_t)b * C_ + c) * NT + qbase + 8 * h;
    v8f cont = *(const v8f*)(content + base);
    v8f mean = accM[ct] * rsv;
    v8f m2   = accS[ct] * rsv;
    v8f o;
#pragma unroll
    for (int r = 0; r < 8; ++r) {
      float sd = sqrtf(fmaxf(m2[r] - mean[r] * mean[r], 0.f));
      o[r] = sd * ((cont[r] - mu_) * rs_) + mean[r];
    }
    *(v8f*)(out + base) = o;
  }
}

// ---------------------------------------------------------------- launcher

extern "C" void kernel_launch(void* const* d_in, const int* in_sizes, int n_in,
                              void* d_out, int out_size, void* d_ws, size_t ws_size,
                              hipStream_t stream) {
  const float* content     = (const float*)d_in[0];
  const float* style       = (const float*)d_in[1];
  const float* content_key = (const float*)d_in[2];
  const float* style_key   = (const float*)d_in[3];
  const float* f_w = (const float*)d_in[4];
  const float* f_b = (const float*)d_in[5];
  const float* g_w = (const float*)d_in[6];
  const float* g_b = (const float*)d_in[7];
  const float* h_w = (const float*)d_in[8];
  const float* h_b = (const float*)d_in[9];
  float* outp = (float*)d_out;

  const size_t BNC = (size_t)B_ * NT * C_;         // 8,388,608 elements
  char* w = (char*)d_ws;
  _Float16* xq  = (_Float16*)w;  w += BNC * 2;     // content_key, token-major f16
  _Float16* xk  = (_Float16*)w;  w += BNC * 2;     // style_key
  _Float16* xv  = (_Float16*)w;  w += BNC * 2;     // style
  _Float16* wf  = (_Float16*)w;  w += (size_t)3 * C_ * C_ * 2;  // f,g,h weights f16
  _Float16* Qh  = (_Float16*)w;  w += BNC * 2;     // [b][t][c]
  _Float16* Kh  = (_Float16*)w;  w += BNC * 2;     // [b][t][c]
  _Float16* Vt  = (_Float16*)w;  w += BNC * 2;     // [b][c][t]
  _Float16* V2t = (_Float16*)w;  w += BNC * 2;     // [b][c][t]
  float* muv    = (float*)w;     w += (size_t)B_ * C_ * 4;
  float* rstdv  = (float*)w;     w += (size_t)B_ * C_ * 4;

  // 1) transpose + convert inputs to token-major f16
  dim3 tb(32, 8), tg(NT / 32, C_ / 32, B_);
  k_transpose_cvt<<<tg, tb, 0, stream>>>(content_key, xq);
  k_transpose_cvt<<<tg, tb, 0, stream>>>(style_key,   xk);
  k_transpose_cvt<<<tg, tb, 0, stream>>>(style,       xv);

  // 2) convert weights
  const int WN = C_ * C_;
  k_cvt_w<<<(WN + 255) / 256, 256, 0, stream>>>(f_w, wf,          WN);
  k_cvt_w<<<(WN + 255) / 256, 256, 0, stream>>>(g_w, wf + WN,     WN);
  k_cvt_w<<<(WN + 255) / 256, 256, 0, stream>>>(h_w, wf + 2 * WN, WN);

  // 3) instance-norm statistics of content
  k_inorm_stats<<<B_ * C_, 256, 0, stream>>>(content, muv, rstdv);

  // 4) projections (WMMA GEMMs)
  dim3 pg(B_ * 256 / 4, C_ / 128);
  k_proj<<<pg, 128, 0, stream>>>(xq, wf,          f_b, Qh, nullptr, nullptr, 0);
  k_proj<<<pg, 128, 0, stream>>>(xk, wf + WN,     g_b, Kh, nullptr, nullptr, 0);
  k_proj<<<pg, 128, 0, stream>>>(xv, wf + 2 * WN, h_b, nullptr, Vt, V2t,    1);

  // 5) fused attention + AdaAttN epilogue (TDM-staged, double-buffered)
  dim3 ag(B_ * 256 / 4, 2);
  k_attn<<<ag, 128, 0, stream>>>(Qh, Kh, Vt, V2t, content, muv, rstdv, outp);
}
